// HeteroGATLayer_52166672777258
// MI455X (gfx1250) — matrix operand order repaired and checked by reading, not compiled, not run on hardware
//
#include <hip/hip_runtime.h>
#include <math.h>

#define NN    100000   // N_P == N_A
#define DD    64
#define EE    1000000
#define SLOPE 0.2f
#define WT_STRIDE 72   // padded f16 row stride (144 B, 16B-aligned) for vector ds loads

typedef __attribute__((ext_vector_type(16))) _Float16 v16h;
typedef __attribute__((ext_vector_type(8)))  float    v8f;
typedef __attribute__((ext_vector_type(4)))  unsigned u32x4;
typedef __attribute__((ext_vector_type(8)))  int      i32x8;
typedef __attribute__((ext_vector_type(4)))  int      i32x4;

// ---------------------------------------------------------------------------
// Generic 64-wide GEMM:  Y[n x 64] = X[n x 64] @ W[64 x 64] + bias
// mode 0: plain bias epilogue
// mode 1: bias + hard gumbel binarize epilogue (straight-through forward == hard)
// W is staged into LDS by the Tensor Data Mover (wave 0 issues TENSOR_LOAD_TO_LDS,
// waits TENSORcnt, block barrier), then converted once to a transposed, padded
// f16 copy so each lane's B fragment is a contiguous 32-byte LDS run.
// One wave computes a 16x16 output tile via two v_wmma_f32_16x16x32_f16 (K=64).
// ---------------------------------------------------------------------------
__global__ __launch_bounds__(256) void gemm64_wmma(
    const float* __restrict__ X, const float* __restrict__ W,
    const float* __restrict__ bias,
    const float* __restrict__ binw, const float* __restrict__ binb,
    const float* __restrict__ U,
    float* __restrict__ Y, int n, int mode)
{
  __shared__ float    sWf[DD * DD];          // 16 KB TDM destination (row-major f32)
  __shared__ _Float16 sWt[DD * WT_STRIDE];   // transposed f16, padded rows

#if __has_builtin(__builtin_amdgcn_tensor_load_to_lds)
  if (threadIdx.x < 32) {                    // one wave drives the TDM
    const unsigned lds_lo = (unsigned)(size_t)(&sWf[0]);
    const unsigned long long ga = (unsigned long long)(size_t)W;
    // D# group 0: count=1 | lds_addr | global_addr[56:0] | type=2 ("image")
    u32x4 g0 = { 1u,
                 lds_lo,
                 (unsigned)(ga & 0xFFFFFFFFu),
                 (unsigned)((unsigned)(ga >> 32) | (2u << 30)) };
    // D# group 1: data_size=2 (4B), tensor 64x64, tile 64x64, dim0_stride=64
    i32x8 g1 = { (int)0x00020000u,        // [17:16] data_size = 2 -> 4-byte elems
                 (int)(64u << 16),        // bits 79:48  tensor_dim0 = 64 (low half)
                 (int)(64u << 16),        // bits 111:80 tensor_dim1 = 64 (low half)
                 (int)(64u << 16),        // bits 127:112 tile_dim0 = 64
                 64,                      // bits 143:128 tile_dim1 = 64 (tile_dim2=0)
                 64,                      // bits 191:160 tensor_dim0_stride = 64
                 (int)(4096u << 16),      // bits 223:208 tensor_dim1_stride[15:0]
                 0 };
    i32x4 gz4 = { 0, 0, 0, 0 };           // groups 2/3 unused for 2D tile
    i32x8 gz8 = { 0, 0, 0, 0, 0, 0, 0, 0 };
    // 6-arg form (clang-23 / therock lane): (g0, g1, g2, g3, extra, cpol)
    __builtin_amdgcn_tensor_load_to_lds(g0, g1, gz4, gz4, gz8, 0);
    __builtin_amdgcn_s_wait_tensorcnt(0);
  }
#else
  for (int i = threadIdx.x; i < DD * DD; i += 256) sWf[i] = W[i];
#endif
  __syncthreads();

  // Transposed, padded f16 copy: sWt[n][k] = W[k][n]; contiguous reads of sWf.
  for (int i = threadIdx.x; i < DD * DD; i += 256) {
    const int k = i >> 6, c = i & 63;
    sWt[c * WT_STRIDE + k] = (_Float16)sWf[i];
  }
  __syncthreads();

  const int wave = threadIdx.x >> 5;
  const int lane = threadIdx.x & 31;
  const int rb_total = (n + 15) >> 4;
  const int gw = blockIdx.x * 8 + wave;          // 8 waves / 256-thread block
  if (gw >= rb_total * 4) return;
  const int row_block = gw >> 2;                 // 16-row block
  const int col = (gw & 3) << 4;                 // 16-col tile (4 tiles cover 64)
  const int lh = lane & 15;
  const int hi = lane >> 4;                      // wave32 half select
  int row = row_block * 16 + lh;
  if (row >= n) row = n - 1;
  const float* xrow = X + (size_t)row * DD;

  v8f acc = {};
  #pragma unroll
  for (int kb = 0; kb < DD; kb += 32) {
    // A k order chosen here (not the ISA table order): lane's K set = {kb+16*hi+0..15} (matches B)
    const float4* xv = (const float4*)(xrow + kb + 16 * hi);
    const float4 a0 = xv[0], a1 = xv[1], a2 = xv[2], a3 = xv[3];
    v16h a;
    a[0]=(_Float16)a0.x; a[1]=(_Float16)a0.y; a[2]=(_Float16)a0.z; a[3]=(_Float16)a0.w;
    a[4]=(_Float16)a1.x; a[5]=(_Float16)a1.y; a[6]=(_Float16)a1.z; a[7]=(_Float16)a1.w;
    a[8]=(_Float16)a2.x; a[9]=(_Float16)a2.y; a[10]=(_Float16)a2.z; a[11]=(_Float16)a2.w;
    a[12]=(_Float16)a3.x; a[13]=(_Float16)a3.y; a[14]=(_Float16)a3.z; a[15]=(_Float16)a3.w;
    // B k order chosen here (same as A): lanes lo K=kb+0..15, lanes hi K=kb+16..31; N=col+lh.
    // Transposed+padded LDS copy makes this one contiguous 32B (16B-aligned) run.
    const v16h b = *(const v16h*)(&sWt[(col + lh) * WT_STRIDE + kb + 16 * hi]);
    acc = __builtin_amdgcn_wmma_f32_16x16x32_f16(false, a, false, b,
                                                 (short)0, acc, false, false);
  }

  // C/D layout: lanes 0-15 -> M = vgpr, lanes 16-31 -> M = vgpr+8; N = lh
  const int c = col + lh;
  const float bc = bias[c];
  float w0 = 0.f, w1 = 0.f, b0 = 0.f, b1 = 0.f;
  if (mode == 1) { w0 = binw[0]; w1 = binw[1]; b0 = binb[0]; b1 = binb[1]; }
  #pragma unroll
  for (int i = 0; i < 8; ++i) {
    const int r = row_block * 16 + i + 8 * hi;
    if (r >= n) continue;
    float v = acc[i] + bc;
    if (mode == 1) {
      const size_t uo = ((size_t)r * DD + c) * 2;
      const float u0 = U[uo + 0], u1 = U[uo + 1];
      const float g0 = -logf(-logf(u0 + 1e-20f) + 1e-20f);
      const float g1 = -logf(-logf(u1 + 1e-20f) + 1e-20f);
      // argmax over 2 logits+gumbel; jnp.argmax picks index 0 on tie -> strict >
      v = ((v * w1 + b1 + g1) > (v * w0 + b0 + g0)) ? 1.0f : 0.0f;
    }
    Y[(size_t)r * DD + c] = v;
  }
}

// ---------------------------------------------------------------------------
// Monotonic float <-> uint32 ordering for atomicMax-based segment max
// ---------------------------------------------------------------------------
__device__ __forceinline__ unsigned fenc(float f) {
  unsigned u = __float_as_uint(f);
  return (u & 0x80000000u) ? ~u : (u | 0x80000000u);
}
__device__ __forceinline__ float fdec(unsigned u) {
  return (u & 0x80000000u) ? __uint_as_float(u & 0x7FFFFFFFu)
                           : __uint_as_float(~u);
}

// s[i] = dot(X[i, 0:64], w)
__global__ __launch_bounds__(256) void dot64_kernel(
    const float* __restrict__ X, const float* __restrict__ w,
    float* __restrict__ out, int n)
{
  int i = blockIdx.x * blockDim.x + threadIdx.x;
  if (i >= n) return;
  const float4* x4 = (const float4*)(X + (size_t)i * DD);
  float s = 0.f;
  #pragma unroll
  for (int q = 0; q < 16; ++q) {
    float4 v = x4[q];
    s += v.x * w[4*q] + v.y * w[4*q+1] + v.z * w[4*q+2] + v.w * w[4*q+3];
  }
  out[i] = s;
}

// e = leaky_relu(s_src[src]+s_dst[dst]); store; segment max via ordered-uint atomicMax
__global__ __launch_bounds__(256) void edge_score_max_kernel(
    const float* __restrict__ sSrc, const float* __restrict__ sDst,
    const int* __restrict__ src, const int* __restrict__ dst,
    float* __restrict__ esc, unsigned* __restrict__ mmax, int ne)
{
  int e = blockIdx.x * blockDim.x + threadIdx.x;
  if (e >= ne) return;
  int d = dst[e];
  float v = sSrc[src[e]] + sDst[d];
  v = (v > 0.f) ? v : SLOPE * v;
  esc[e] = v;
  atomicMax(&mmax[d], fenc(v));
}

// m = isfinite(decode(mmax)) ? m : 0   (init 0u decodes to a NaN pattern -> 0)
__global__ __launch_bounds__(256) void finalize_max_kernel(
    const unsigned* __restrict__ mmax, float* __restrict__ mfin, int n)
{
  int i = blockIdx.x * blockDim.x + threadIdx.x;
  if (i >= n) return;
  float m = fdec(mmax[i]);
  unsigned eb = __float_as_uint(m) & 0x7F800000u;
  mfin[i] = (eb != 0x7F800000u) ? m : 0.f;
}

// ex = exp(e - m[dst]); store in place; segment sum of ex
__global__ __launch_bounds__(256) void edge_exp_kernel(
    float* __restrict__ esc, const float* __restrict__ mfin,
    const int* __restrict__ dst, float* __restrict__ den, int ne)
{
  int e = blockIdx.x * blockDim.x + threadIdx.x;
  if (e >= ne) return;
  int d = dst[e];
  float ex = __expf(esc[e] - mfin[d]);
  esc[e] = ex;
  atomicAdd(&den[d], ex);
}

__global__ __launch_bounds__(256) void dinv_kernel(
    const float* __restrict__ den, float* __restrict__ dinv, int n)
{
  int i = blockIdx.x * blockDim.x + threadIdx.x;
  if (i >= n) return;
  float d = den[i];
  dinv[i] = 1.0f / ((d > 0.f) ? d : 1.0f);
}

// acc[dst] += alpha * z[src]; 64 threads per edge (coalesced 256B gather/atomic)
__global__ __launch_bounds__(256) void edge_scatter_kernel(
    const float* __restrict__ esc, const float* __restrict__ dinv,
    const float* __restrict__ z,
    const int* __restrict__ src, const int* __restrict__ dst,
    float* __restrict__ acc, int ne)
{
  int t = blockIdx.x * 256 + threadIdx.x;
  int e = t >> 6;
  int k = t & 63;
  if (e >= ne) return;
  int s = src[e], d = dst[e];
  float alpha = esc[e] * dinv[d];
  atomicAdd(&acc[(size_t)d * DD + k], alpha * z[(size_t)s * DD + k]);
}

__global__ __launch_bounds__(256) void relu_kernel(float* __restrict__ y, size_t n)
{
  size_t i = (size_t)blockIdx.x * blockDim.x + threadIdx.x;
  if (i >= n) return;
  y[i] = fmaxf(y[i], 0.f);
}

// ---------------------------------------------------------------------------
extern "C" void kernel_launch(void* const* d_in, const int* in_sizes, int n_in,
                              void* d_out, int out_size, void* d_ws, size_t ws_size,
                              hipStream_t stream)
{
  (void)in_sizes; (void)n_in; (void)out_size; (void)ws_size;

  const float* feat_P = (const float*)d_in[0];
  const float* feat_A = (const float*)d_in[1];
  const float* fcP_w  = (const float*)d_in[2];  const float* fcP_b  = (const float*)d_in[3];
  const float* fcA_w  = (const float*)d_in[4];  const float* fcA_b  = (const float*)d_in[5];
  const float* encP_w = (const float*)d_in[6];  const float* encP_b = (const float*)d_in[7];
  const float* encA_w = (const float*)d_in[8];  const float* encA_b = (const float*)d_in[9];
  const float* bin_w  = (const float*)d_in[10]; const float* bin_b  = (const float*)d_in[11];
  const float* decP_w = (const float*)d_in[12]; const float* decP_b = (const float*)d_in[13];
  const float* decA_w = (const float*)d_in[14]; const float* decA_b = (const float*)d_in[15];
  const float* aw_p2p = (const float*)d_in[16];
  const float* aw_p2a = (const float*)d_in[17];
  const float* aw_a2p = (const float*)d_in[18];
  const float* aw_a2a = (const float*)d_in[19];
  const float* u_P    = (const float*)d_in[20];
  const float* u_A    = (const float*)d_in[21];
  const int* p2p_src = (const int*)d_in[22]; const int* p2p_dst = (const int*)d_in[23];
  const int* p2a_src = (const int*)d_in[24]; const int* p2a_dst = (const int*)d_in[25];
  const int* a2p_src = (const int*)d_in[26]; const int* a2p_dst = (const int*)d_in[27];
  const int* a2a_src = (const int*)d_in[28]; const int* a2a_dst = (const int*)d_in[29];

  const size_t NF = (size_t)NN * DD;
  // d_out doubles as the accumulator: acc = Wh (GEMM), += recv (atomics), relu in place
  float* accP = (float*)d_out;
  float* accA = accP + NF;

  float* ws   = (float*)d_ws;
  float* mPP  = ws;             // msg_P decoded to P
  float* mPA  = ws + 1 * NF;    // msg_P decoded to A
  float* mAP  = ws + 2 * NF;    // msg_A decoded to P
  float* mAA  = ws + 3 * NF;    // msg_A decoded to A
  float* msgP = ws + 4 * NF;    // binary messages (reused as scratch after dec GEMMs)
  float* msgA = ws + 5 * NF;

  const int gemm_blocks = (((NN + 15) / 16) * 4 + 7) / 8;
  // Wh_P / Wh_A directly into the output accumulators
  gemm64_wmma<<<gemm_blocks, 256, 0, stream>>>(feat_P, fcP_w,  fcP_b,  bin_w, bin_b, u_P, accP, NN, 0);
  gemm64_wmma<<<gemm_blocks, 256, 0, stream>>>(feat_A, fcA_w,  fcA_b,  bin_w, bin_b, u_A, accA, NN, 0);
  // encoder + fused gumbel hard-binarize
  gemm64_wmma<<<gemm_blocks, 256, 0, stream>>>(feat_P, encP_w, encP_b, bin_w, bin_b, u_P, msgP, NN, 1);
  gemm64_wmma<<<gemm_blocks, 256, 0, stream>>>(feat_A, encA_w, encA_b, bin_w, bin_b, u_A, msgA, NN, 1);
  // decode binary messages per destination type
  gemm64_wmma<<<gemm_blocks, 256, 0, stream>>>(msgP, decP_w, decP_b, bin_w, bin_b, u_P, mPP, NN, 0);
  gemm64_wmma<<<gemm_blocks, 256, 0, stream>>>(msgP, decA_w, decA_b, bin_w, bin_b, u_P, mPA, NN, 0);
  gemm64_wmma<<<gemm_blocks, 256, 0, stream>>>(msgA, decP_w, decP_b, bin_w, bin_b, u_A, mAP, NN, 0);
  gemm64_wmma<<<gemm_blocks, 256, 0, stream>>>(msgA, decA_w, decA_b, bin_w, bin_b, u_A, mAA, NN, 0);

  // scratch: msgP region -> per-edge score/exp buffers (4 * 1M <= 6.4M floats)
  //          msgA region -> per-node scalars (24 * 100k <= 6.4M floats)
  float*    esc  = msgP;
  float*    scr  = msgA;
  float*    sSrc = scr;                                // 4*NN
  float*    sDst = scr + 4 * (size_t)NN;               // 4*NN
  unsigned* mmax = (unsigned*)(scr + 8 * (size_t)NN);  // 4*NN
  float*    den  = scr + 12 * (size_t)NN;              // 4*NN
  float*    mfin = scr + 16 * (size_t)NN;              // 4*NN
  float*    dinv = scr + 20 * (size_t)NN;              // 4*NN
  (void)hipMemsetAsync(scr + 8 * (size_t)NN, 0, 8 * (size_t)NN * sizeof(float), stream);

  // relation order: p2p, a2p (dst P), p2a, a2a (dst A)
  const float* zs[4]  = {mPP, mAP, mPA, mAA};
  const float* aws[4] = {aw_p2p, aw_a2p, aw_p2a, aw_a2a};
  const int* srcs[4]  = {p2p_src, a2p_src, p2a_src, a2a_src};
  const int* dsts[4]  = {p2p_dst, a2p_dst, p2a_dst, a2a_dst};
  float* accs[4]      = {accP, accP, accA, accA};

  const int nodeBlocks = (NN + 255) / 256;
  const int edgeBlocks = (EE + 255) / 256;
  const int scatBlocks = (int)(((size_t)EE * 64 + 255) / 256);

  // node scalar projections (must read acc == Wh BEFORE any scatter mutates it)
  for (int r = 0; r < 4; ++r) {
    dot64_kernel<<<nodeBlocks, 256, 0, stream>>>(zs[r],   aws[r],      sSrc + r * (size_t)NN, NN);
    dot64_kernel<<<nodeBlocks, 256, 0, stream>>>(accs[r], aws[r] + DD, sDst + r * (size_t)NN, NN);
  }
  for (int r = 0; r < 4; ++r)
    edge_score_max_kernel<<<edgeBlocks, 256, 0, stream>>>(
        sSrc + r * (size_t)NN, sDst + r * (size_t)NN, srcs[r], dsts[r],
        esc + r * (size_t)EE, mmax + r * (size_t)NN, EE);
  for (int r = 0; r < 4; ++r)
    finalize_max_kernel<<<nodeBlocks, 256, 0, stream>>>(
        mmax + r * (size_t)NN, mfin + r * (size_t)NN, NN);
  for (int r = 0; r < 4; ++r)
    edge_exp_kernel<<<edgeBlocks, 256, 0, stream>>>(
        esc + r * (size_t)EE, mfin + r * (size_t)NN, dsts[r], den + r * (size_t)NN, EE);
  for (int r = 0; r < 4; ++r)
    dinv_kernel<<<nodeBlocks, 256, 0, stream>>>(
        den + r * (size_t)NN, dinv + r * (size_t)NN, NN);
  for (int r = 0; r < 4; ++r)
    edge_scatter_kernel<<<scatBlocks, 256, 0, stream>>>(
        esc + r * (size_t)EE, dinv + r * (size_t)NN, zs[r], srcs[r], dsts[r],
        accs[r], EE);

  const size_t tot = 2 * NF;
  relu_kernel<<<(int)((tot + 255) / 256), 256, 0, stream>>>((float*)d_out, tot);
}
